// CNCNeRF_11587821765185
// MI455X (gfx1250) — matrix operand order repaired and checked by reading, not compiled, not run on hardware
//
#include <hip/hip_runtime.h>
#include <hip/hip_bf16.h>
#include <math.h>

typedef __attribute__((ext_vector_type(16))) _Float16 v16h;
typedef __attribute__((ext_vector_type(8)))  _Float16 v8h;
typedef __attribute__((ext_vector_type(8)))  float    v8f;

#define NUM_LEVELS   12
#define TABLE_SIZE   (1u << 19)
#define FEAT_DIM     8
#define PTS_PER_BLK  128
#define THREADS      256
#define FSTRIDE      136   // halfs per feature row: 128 used + pad (272B, 16B aligned, 4-bank rotate)

__device__ __forceinline__ v8f wmma_f16(v16h a, v16h b, v8f c) {
  // D = A(16x32 f16) * B(32x16 f16) + C(16x16 f32)
  return __builtin_amdgcn_wmma_f32_16x16x32_f16(false, a, false, b, (short)0, c, false, false);
}

// A-fragment: elements 0..7 = p0[0..7] (K=khalf*8..+7), 8..15 = p0[16..23] (K=16+khalf*8..)
__device__ __forceinline__ v16h ld_a_frag(const _Float16* p0) {
  union { v8h h[2]; v16h v; } u;
  u.h[0] = *(const v8h*)(p0);
  u.h[1] = *(const v8h*)(p0 + 16);
  return u.v;
}

// B-fragment: 32 contiguous bytes pre-swizzled in LDS
__device__ __forceinline__ v16h ld_b_frag(const _Float16* p) {
  union { v8h h[2]; v16h v; } u;
  u.h[0] = *(const v8h*)(p);
  u.h[1] = *(const v8h*)(p + 8);
  return u.v;
}

__global__ __launch_bounds__(THREADS)
void nerf_hash_mlp_fused(const float* __restrict__ coords,
                         const float* __restrict__ vdirs,
                         const float* __restrict__ tables,
                         const float* __restrict__ w1, const float* __restrict__ b1,
                         const float* __restrict__ w2, const float* __restrict__ b2,
                         const float* __restrict__ w3, const float* __restrict__ b3,
                         float* __restrict__ out, int N)
{
  __shared__ __align__(32) _Float16 sFeat[PTS_PER_BLK * FSTRIDE]; // 34816 B
  __shared__ __align__(32) _Float16 sW1[8192];                    // 16384 B, frag order
  __shared__ __align__(32) _Float16 sW2[4096];                    //  8192 B
  __shared__ __align__(32) _Float16 sW3[1024];                    //  2048 B

  const int t = threadIdx.x;
  const int blockBase = blockIdx.x * PTS_PER_BLK;

  // ---------------- stage weights into WMMA B-fragment order ----------------
  // frag index f = ((kk*NT + nt)*32 + lane)*16 + e ; k = kk*32 + (lane>>4)*8 + (e&7) + (e>>3)*16 ; n = nt*16 + (lane&15)
  for (int f = t; f < 8192; f += THREADS) {               // W1: 99(->128) x 64
    int e = f & 15, lane = (f >> 4) & 31, tile = f >> 9;  // tile = kk*4 + nt
    int k = (tile >> 2) * 32 + ((lane >> 4) << 3) + (e & 7) + ((e >> 3) << 4);
    int n = (tile & 3) * 16 + (lane & 15);
    sW1[f] = (_Float16)((k < 99) ? w1[k * 64 + n] : 0.f);
  }
  for (int f = t; f < 4096; f += THREADS) {               // W2: 64 x 64
    int e = f & 15, lane = (f >> 4) & 31, tile = f >> 9;
    int k = (tile >> 2) * 32 + ((lane >> 4) << 3) + (e & 7) + ((e >> 3) << 4);
    int n = (tile & 3) * 16 + (lane & 15);
    sW2[f] = (_Float16)w2[k * 64 + n];
  }
  for (int f = t; f < 1024; f += THREADS) {               // W3: 64 x 4(->16)
    int e = f & 15, lane = (f >> 4) & 31, kk = f >> 9;
    int k = kk * 32 + ((lane >> 4) << 3) + (e & 7) + ((e >> 3) << 4);
    int n = lane & 15;
    sW3[f] = (_Float16)((n < 4) ? w3[k * 4 + n] : 0.f);
  }

  // ---------------- multi-resolution hash encode: 2 threads / point ----------------
  {
    const float res_tab[NUM_LEVELS] = {16.f, 21.f, 30.f, 41.f, 56.f, 77.f,
                                       105.f, 145.f, 198.f, 272.f, 373.f, 512.f};
    const int p = t >> 1, half = t & 1;
    int point = blockBase + p;
    if (point >= N) point = N - 1;                        // clamp (N is a multiple of 128 anyway)
    const float cx = coords[point * 3 + 0];
    const float cy = coords[point * 3 + 1];
    const float cz = coords[point * 3 + 2];
    _Float16* frow = &sFeat[p * FSTRIDE];

    for (int li = 0; li < 6; ++li) {
      const int l = half * 6 + li;
      const float res = res_tab[l];
      const float sx = cx * res, sy = cy * res, sz = cz * res;
      const float px = floorf(sx), py = floorf(sy), pz = floorf(sz);
      const float fx = sx - px, fy = sy - py, fz = sz - pz;
      const unsigned ix = (unsigned)px, iy = (unsigned)py, iz = (unsigned)pz;
      const float* tab = tables + (size_t)l * (size_t)TABLE_SIZE * FEAT_DIM;
      float a0 = 0.f, a1 = 0.f, a2 = 0.f, a3 = 0.f, a4 = 0.f, a5 = 0.f, a6 = 0.f, a7 = 0.f;
      #pragma unroll
      for (int j = 0; j < 8; ++j) {
        const unsigned gx = ix + ((j >> 2) & 1);
        const unsigned gy = iy + ((j >> 1) & 1);
        const unsigned gz = iz + (j & 1);
        const unsigned idx = (gx ^ (gy * 2654435761u) ^ (gz * 805459861u)) & (TABLE_SIZE - 1u);
        const float4* fp = (const float4*)(tab + (size_t)idx * FEAT_DIM);
        const float4 f0 = fp[0];
        const float4 f1 = fp[1];
        const float w = ((j & 4) ? fx : 1.f - fx) *
                        ((j & 2) ? fy : 1.f - fy) *
                        ((j & 1) ? fz : 1.f - fz);
        a0 = fmaf(w, f0.x, a0); a1 = fmaf(w, f0.y, a1);
        a2 = fmaf(w, f0.z, a2); a3 = fmaf(w, f0.w, a3);
        a4 = fmaf(w, f1.x, a4); a5 = fmaf(w, f1.y, a5);
        a6 = fmaf(w, f1.z, a6); a7 = fmaf(w, f1.w, a7);
      }
      _Float16* fo = frow + l * FEAT_DIM;
      fo[0] = (_Float16)a0; fo[1] = (_Float16)a1; fo[2] = (_Float16)a2; fo[3] = (_Float16)a3;
      fo[4] = (_Float16)a4; fo[5] = (_Float16)a5; fo[6] = (_Float16)a6; fo[7] = (_Float16)a7;
    }
    if (half == 0) {                                       // view dirs -> k = 96..98
      frow[96] = (_Float16)vdirs[point * 3 + 0];
      frow[97] = (_Float16)vdirs[point * 3 + 1];
      frow[98] = (_Float16)vdirs[point * 3 + 2];
    } else {                                               // zero pad k = 99..127
      for (int k = 99; k < 128; ++k) frow[k] = (_Float16)0.f;
    }
  }
  __syncthreads();

  // ---------------- fused MLP: one 16-point tile per wave ----------------
  const int lane   = t & 31;
  const int wid    = t >> 5;
  const int rowBase = wid * 16;
  const int nlo    = lane & 15;       // A: row within tile; B/C/D: column
  const int khalf  = lane >> 4;
  const _Float16* fr = &sFeat[(rowBase + nlo) * FSTRIDE]; // this lane's A row
  _Float16* const sRowBase = &sFeat[rowBase * FSTRIDE];

  // layer 1: [16x128] x [128x64]
  v16h A1[4];
  #pragma unroll
  for (int kk = 0; kk < 4; ++kk) A1[kk] = ld_a_frag(fr + kk * 32 + khalf * 8);
  #pragma unroll
  for (int nt = 0; nt < 4; ++nt) {
    const float bn = b1[nt * 16 + nlo];
    v8f c; 
    #pragma unroll
    for (int r = 0; r < 8; ++r) c[r] = bn;
    #pragma unroll
    for (int kk = 0; kk < 4; ++kk)
      c = wmma_f16(A1[kk], ld_b_frag(&sW1[((kk * 4 + nt) * 32 + lane) * 16]), c);
    #pragma unroll
    for (int r = 0; r < 8; ++r) {                          // ReLU, h1 -> LDS k=0..63
      const int m = khalf * 8 + r;
      sRowBase[m * FSTRIDE + nt * 16 + nlo] = (_Float16)fmaxf(c[r], 0.f);
    }
  }
  asm volatile("s_wait_dscnt 0x0" ::: "memory");

  // layer 2: [16x64] x [64x64]
  v16h A2[2];
  #pragma unroll
  for (int kk = 0; kk < 2; ++kk) A2[kk] = ld_a_frag(fr + kk * 32 + khalf * 8);
  #pragma unroll
  for (int nt = 0; nt < 4; ++nt) {
    const float bn = b2[nt * 16 + nlo];
    v8f c;
    #pragma unroll
    for (int r = 0; r < 8; ++r) c[r] = bn;
    #pragma unroll
    for (int kk = 0; kk < 2; ++kk)
      c = wmma_f16(A2[kk], ld_b_frag(&sW2[((kk * 4 + nt) * 32 + lane) * 16]), c);
    #pragma unroll
    for (int r = 0; r < 8; ++r) {                          // ReLU, h2 -> LDS k=64..127
      const int m = khalf * 8 + r;
      sRowBase[m * FSTRIDE + 64 + nt * 16 + nlo] = (_Float16)fmaxf(c[r], 0.f);
    }
  }
  asm volatile("s_wait_dscnt 0x0" ::: "memory");

  // layer 3: [16x64] x [64x16] (only n=0..3 valid)
  v16h A3[2];
  #pragma unroll
  for (int kk = 0; kk < 2; ++kk) A3[kk] = ld_a_frag(fr + 64 + kk * 32 + khalf * 8);
  {
    const float bn = (nlo < 4) ? b3[nlo] : 0.f;
    v8f c;
    #pragma unroll
    for (int r = 0; r < 8; ++r) c[r] = bn;
    c = wmma_f16(A3[0], ld_b_frag(&sW3[lane * 16]), c);
    c = wmma_f16(A3[1], ld_b_frag(&sW3[512 + lane * 16]), c);

    if (nlo < 4) {
      float* const rgb = out;            // [N,3]
      float* const den = out + 3 * (size_t)N;
      #pragma unroll
      for (int r = 0; r < 8; ++r) {
        const int point = blockBase + rowBase + khalf * 8 + r;
        if (point < N) {
          const float v = c[r];
          if (nlo < 3) rgb[(size_t)point * 3 + nlo] = 1.f / (1.f + expf(-v));
          else         den[point] = fmaxf(v, 0.f);
        }
      }
    }
  }
}

extern "C" void kernel_launch(void* const* d_in, const int* in_sizes, int n_in,
                              void* d_out, int out_size, void* d_ws, size_t ws_size,
                              hipStream_t stream) {
  const float* coords = (const float*)d_in[0];
  const float* vdirs  = (const float*)d_in[1];
  const float* tables = (const float*)d_in[2];
  const float* w1 = (const float*)d_in[3];
  const float* b1 = (const float*)d_in[4];
  const float* w2 = (const float*)d_in[5];
  const float* b2 = (const float*)d_in[6];
  const float* w3 = (const float*)d_in[7];
  const float* b3 = (const float*)d_in[8];
  float* out = (float*)d_out;

  const int N = in_sizes[0] / 3;
  const int grid = (N + PTS_PER_BLK - 1) / PTS_PER_BLK;
  hipLaunchKernelGGL(nerf_hash_mlp_fused, dim3(grid), dim3(THREADS), 0, stream,
                     coords, vdirs, tables, w1, b1, w2, b2, w3, b3, out, N);
}
